// NNet_18159121727994
// MI455X (gfx1250) — matrix operand order repaired and checked by reading, not compiled, not run on hardware
//
#include <hip/hip_runtime.h>
#include <math.h>

// ---------------------------------------------------------------------------
// SpectralNet-style 3D denoiser for MI455X (gfx1250, wave32, WMMA).
//
// Activations: f16, channels-last [D][H][W][C], C padded to multiple of 16,
// padding zero-filled. Every conv -> one implicit-GEMM WMMA engine:
//   M = cout (blocked by 16), N = 16 contiguous W voxels,
//   K = 32 = 2 kernel taps x 16 input channels per v_wmma_f32_16x16x32_f16.
// LDS holds: packed per-tap offset table (division-free inner loop), bias,
// and weights pre-swizzled into A fragments (lane half = tap, element = input channel).
// Out-of-bounds taps read a zeroed global page (pointer select, no masking).
// Inner loop is software-pipelined one K-step ahead so the WMMA consumes
// registers loaded a full iteration earlier (partial s_wait, loads in flight).
//
// Input flattening (jax tree_flatten, dict keys sorted):
//   0..5 dec[0..2](b,w) | 6..53 enc[0..2]{p1.a,p1.b,p2.a,p2.b,p3.a,p3.b,x1,x2}(b,w)
//   54,55 feat | 56,57 nconv | 58,59 recon | 60..67 spec1 c1..c4 | 68..75 spec2
//   76..81 spec3 c1..c3 | 82 x [1,1,31,96,96] f32
// Output: x - recon(...) -> 285696 f32.
// ---------------------------------------------------------------------------

typedef __attribute__((ext_vector_type(8)))  _Float16 v8h;
typedef __attribute__((ext_vector_type(16))) _Float16 v16h;
typedef __attribute__((ext_vector_type(8)))  float    v8f;

#define ACT_NONE 0
#define ACT_LRELU 1
#define ACT_RELU 2
#define ACT_SIGMOID 3

struct ConvP {
  const _Float16* in;       // f16 channels-last input
  const float* w;           // f32 weights (cout,cin,kd,kh,kw)
  const float* bias;        // f32 bias (cout)
  const _Float16* res;      // optional f16 residual, same layout as out
  const float* resf;        // optional f32 residual (1ch) for f32-out path
  const _Float16* zp;       // 32B zeroed global page
  _Float16* out;            // f16 channels-last output
  float* outf;              // f32 output (1ch compact) if non-null
  int cin, cout;
  int inD, inH, inW, outD, outH, outW;
  int kd, kh, kw, sd, sh, sw, pd, ph, pw, dd, dh, dw;
  int act;
  int inCtot, inChOff, outCtot, outChOff;
  float resScale, outScale;
  int ntilesW, mbBlocks, cbBlocks, npairs, ntaps, khw;
  int rows, rowBlocks, units;
};

// LDS layout: [0,128)=tap table (int), [128,256)=bias (float32 x32),
//             [256,...)=A fragments.
#define LDS_A_OFF 256

__global__ __launch_bounds__(128) void conv3d_wmma_kernel(ConvP p) {
  extern __shared__ char smem[];
  int*      tapTab  = (int*)smem;
  float*    ldsBias = (float*)(smem + 128);
  _Float16* ldsA    = (_Float16*)(smem + LDS_A_OFF);

  const int tid  = threadIdx.x;
  const int wave = tid >> 5;
  const int lane = tid & 31;
  const int n    = lane & 15;
  const int hsel = lane >> 4;

  // ---- stage tap-offset table: packed (zd*dd-pd, zh*dh-ph, zw*dw-pw) ----
  for (int t = tid; t < 2 * p.npairs; t += 128) {
    int pk = 0x3FFFFFFF;                       // sentinel: +511 offsets -> OOB
    if (t < p.ntaps) {
      const int zd = t / p.khw;
      const int r2 = t - zd * p.khw;
      const int zh = r2 / p.kw;
      const int zw = r2 - zh * p.kw;
      const int oD = zd * p.dd - p.pd;
      const int oH = zh * p.dh - p.ph;
      const int oW = zw * p.dw - p.pw;
      pk = ((oD + 512) << 20) | ((oH + 512) << 10) | (oW + 512);
    }
    tapTab[t] = pk;
  }
  // ---- stage bias ----
  if (tid < 32) ldsBias[tid] = (tid < p.cout) ? p.bias[tid] : 0.0f;

  // ---- stage A fragments (weights), f32 -> f16, WMMA-swizzled, div-free ----
  {
    const int nfrag = p.mbBlocks * p.cbBlocks * p.npairs;
    int tpp = 0, cb = 0, mb = 0;
    for (int fr = 0; fr < nfrag; ++fr) {
#pragma unroll
      for (int k = 0; k < 4; ++k) {
        const int within = tid * 4 + k;        // 0..511
        const int e  = within & 15;            // 2*pp + j
        const int ln = within >> 4;            // lane slot
        const int pp = e >> 1, j = e & 1;
        const int hs = ln >> 4;
        const int tap = tpp * 2 + hs;
        const int kbase = ((pp < 4) ? 2 * pp : 2 * pp + 8) + 8 * hs;
        const int c = cb * 16 + e;
        const int m = mb * 16 + (ln & 15);
        const bool ok = (tap < p.ntaps) && (c < p.cin) && (m < p.cout);
        const int wi = ok ? (m * p.cin + c) * p.ntaps + tap : 0;
        const float wv = p.w[wi];
        ldsA[fr * 512 + within] = (_Float16)(ok ? wv : 0.0f);
      }
      if (++tpp == p.npairs) { tpp = 0; if (++cb == p.cbBlocks) { cb = 0; ++mb; } }
    }
  }
  __syncthreads();

  const int blk = blockIdx.x;
  const int od  = blk / p.rowBlocks;
  const int oh0 = (blk - od * p.rowBlocks) * p.rows;
  const int odsd = od * p.sd;

  for (int u = wave; u < p.units; u += 4) {            // wave-uniform loop
    const int mb = u % p.mbBlocks;
    int t = u / p.mbBlocks;
    const int wt = t % p.ntilesW;
    const int oh = oh0 + t / p.ntilesW;
    if (oh >= p.outH) continue;                         // uniform per wave

    const int ow = wt * 16 + n;
    const bool nvalid = ow < p.outW;
    const int ohsh = oh * p.sh;
    const int owsw = nvalid ? ow * p.sw : (1 << 28);    // fold nvalid into OOB

    v8f acc = {};
    for (int cb = 0; cb < p.cbBlocks; ++cb) {
      const _Float16* aptr =
          ldsA + (size_t)(mb * p.cbBlocks + cb) * p.npairs * 512 + lane * 16;
      const int c0 = p.inChOff + cb * 16;

      // B pointer for tap-pair index t (lane half selects the tap)
      auto bAddr = [&](int tp) -> const _Float16* {
        const int pk = tapTab[(tp << 1) | hsel];        // ds_load_b32
        const int iw = owsw + ((pk & 1023) - 512);
        const int ih = ohsh + (((pk >> 10) & 1023) - 512);
        const int id = odsd + ((pk >> 20) - 512);
        const bool inb = ((unsigned)id < (unsigned)p.inD) &&
                         ((unsigned)ih < (unsigned)p.inH) &&
                         ((unsigned)iw < (unsigned)p.inW);
        const int vox = (id * p.inH + ih) * p.inW + iw;
        return inb ? (p.in + (size_t)vox * p.inCtot + c0) : p.zp;
      };

      // pipeline prologue: load step 0
      v16h A = *(const v16h*)aptr;                      // 2x ds_load_b128
      const _Float16* bp = bAddr(0);
      v8h b0 = ((const v8h*)bp)[0];
      v8h b1 = ((const v8h*)bp)[1];

      for (int tpp = 0; tpp < p.npairs; ++tpp) {
        // issue loads for step tpp+1 (clamped at tail) before consuming tpp
        const int tn = (tpp + 1 < p.npairs) ? tpp + 1 : tpp;
        const v16h An = *(const v16h*)(aptr + (size_t)(tn - tpp) * 512);
        const _Float16* bpn = bAddr(tn);
        __builtin_prefetch(bpn, 0, 1);                  // global_prefetch_b8
        const v8h n0 = ((const v8h*)bpn)[0];
        const v8h n1 = ((const v8h*)bpn)[1];

        v16h B;
#pragma unroll
        for (int i = 0; i < 8; ++i) { B[i] = b0[i]; B[i + 8] = b1[i]; }
        acc = __builtin_amdgcn_wmma_f32_16x16x32_f16(
                  false, A, false, B, (short)0, acc, false, false);

        A = An; b0 = n0; b1 = n1;
        aptr += 512;
      }
    }

    // ---- epilogue: bias + activation (f32), residual, f16 16B store
    const int voxO = (od * p.outH + oh) * p.outW + (nvalid ? ow : 0);
    if (p.outf == nullptr) {
      v8h outv;
#pragma unroll
      for (int r = 0; r < 8; ++r) {
        const int m = mb * 16 + r + 8 * hsel;
        float v = acc[r] + ldsBias[m & 31];
        if (p.act == ACT_LRELU)        v = (v > 0.0f) ? v : 0.05f * v;
        else if (p.act == ACT_RELU)    v = fmaxf(v, 0.0f);
        else if (p.act == ACT_SIGMOID) v = 1.0f / (1.0f + expf(-v));
        v = (m < p.cout) ? v : 0.0f;                    // zero-fill padding ch
        outv[r] = (_Float16)v;
      }
      _Float16* optr = p.out + (size_t)voxO * p.outCtot + p.outChOff +
                       mb * 16 + 8 * hsel;
      if (p.res) {
        const v8h rv = *(const v8h*)((const _Float16*)p.res +
                       (size_t)voxO * p.outCtot + p.outChOff + mb * 16 + 8 * hsel);
#pragma unroll
        for (int r = 0; r < 8; ++r)
          outv[r] = (_Float16)(p.resScale * (float)rv[r] +
                               p.outScale * (float)outv[r]);
      }
      if (nvalid) *(v8h*)optr = outv;
    } else {
      // f32 compact output path (final recon conv, cout==1)
#pragma unroll
      for (int r = 0; r < 8; ++r) {
        const int m = mb * 16 + r + 8 * hsel;
        if (m < p.cout && nvalid) {
          float v = acc[r] + ldsBias[m & 31];
          if (p.act == ACT_LRELU) v = (v > 0.0f) ? v : 0.05f * v;
          const size_t oi = (size_t)voxO * p.cout + m;
          if (p.resf) v = p.resScale * p.resf[oi] + p.outScale * v;
          p.outf[oi] = v;
        }
      }
    }
  }
}

// ---- elementwise / scan helpers (f16 channels-last) ------------------------

__global__ void cast_x_kernel(const float* x, _Float16* xh, _Float16* zp, int V) {
  int i = blockIdx.x * blockDim.x + threadIdx.x;
  if (i >= V) return;
  v8h a = {}; v8h z = {};
  a[0] = (_Float16)x[i];
  *(v8h*)(xh + (size_t)i * 16) = a;
  *(v8h*)(xh + (size_t)i * 16 + 8) = z;
  if (i == 0) { *(v8h*)zp = z; *(v8h*)(zp + 8) = z; }
}

// channels 0..m-1 -> tanh (Z), m..2m-1 -> sigmoid (F), in place
__global__ void zf_kernel(_Float16* g, int m, int Ctot, int V) {
  int i = blockIdx.x * blockDim.x + threadIdx.x;
  if (i >= V * 2 * m) return;
  const int vox = i / (2 * m);
  const int c = i - vox * 2 * m;
  const size_t idx = (size_t)vox * Ctot + c;
  const float v = (float)g[idx];
  g[idx] = (_Float16)((c < m) ? tanhf(v) : 1.0f / (1.0f + expf(-v)));
}

// bidirectional QRNN scan over D; zero-fills padding channels of hd/hs
__global__ void qrnn_scan_kernel(const _Float16* g, _Float16* hd, _Float16* hs,
                                 int m, int D, int HW, int gC, int oC) {
  int i = blockIdx.x * blockDim.x + threadIdx.x;
  if (i >= 16 * HW) return;
  const int c = i / HW;
  const int pix = i - c * HW;
  if (c >= m) {
    for (int d = 0; d < D; ++d) {
      const size_t o = (size_t)(d * HW + pix) * oC + c;
      hd[o] = (_Float16)0.0f; hs[o] = (_Float16)0.0f;
    }
    return;
  }
  float h = 0.0f;
  for (int d = 0; d < D; ++d) {
    const size_t gi = (size_t)(d * HW + pix) * gC;
    const float z = (float)g[gi + c], f = (float)g[gi + m + c];
    h = f * h + (1.0f - f) * z;
    hd[(size_t)(d * HW + pix) * oC + c] = (_Float16)h;
  }
  h = 0.0f;
  for (int d = D - 1; d >= 0; --d) {
    const size_t gi = (size_t)(d * HW + pix) * gC;
    const float z = (float)g[gi + c], f = (float)g[gi + m + c];
    h = f * h + (1.0f - f) * z;
    const size_t o = (size_t)(d * HW + pix) * oC + c;
    hd[o] = (_Float16)((float)hd[o] + h);
    hs[o] = (_Float16)((1.0f - f) * z);
  }
}

__global__ void axpy_kernel(_Float16* out, const _Float16* a, const _Float16* b,
                            float s, int n) {
  int i = blockIdx.x * blockDim.x + threadIdx.x;
  if (i < n) out[i] = (_Float16)((float)a[i] + s * (float)b[i]);
}

// nx[vox,c] = x[vox,c] * noise[d,c]
__global__ void mulnoise_kernel(_Float16* out, const _Float16* x,
                                const _Float16* noise, int HW, int n) {
  int i = blockIdx.x * blockDim.x + threadIdx.x;
  if (i >= n) return;
  const int vox = i >> 4;
  const int c = i & 15;
  const int d = vox / HW;
  out[i] = (_Float16)((float)x[i] * (float)noise[d * 16 + c]);
}

// exact rH x rW block mean, channels-last (16 ch)
__global__ void blockmean_kernel(const _Float16* in, _Float16* out,
                                 int iH, int iW, int oH, int oW,
                                 int rH, int rW, int n) {
  int i = blockIdx.x * blockDim.x + threadIdx.x;
  if (i >= n) return;
  int t = i >> 4;
  const int c = i & 15;
  const int ow = t % oW; t /= oW;
  const int oh = t % oH; t /= oH;
  const int d = t;
  float s = 0.0f;
  for (int r = 0; r < rH; ++r)
    for (int q = 0; q < rW; ++q)
      s += (float)in[(size_t)((d * iH + oh * rH + r) * iW + ow * rW + q) * 16 + c];
  out[i] = (_Float16)(s / (float)(rH * rW));
}

// out[d,c] = mean over HW of in[d,:,:,c]
__global__ void meanhw_kernel(const _Float16* in, _Float16* out, int Dn, int HW) {
  int i = blockIdx.x * blockDim.x + threadIdx.x;
  if (i >= Dn * 16) return;
  const int d = i >> 4;
  const int c = i & 15;
  float s = 0.0f;
  for (int k = 0; k < HW; ++k) s += (float)in[(size_t)(d * HW + k) * 16 + c];
  out[i] = (_Float16)(s / (float)HW);
}

// ---- host orchestration ----------------------------------------------------

static inline int cdiv(int a, int b) { return (a + b - 1) / b; }

static const _Float16* g_zp = nullptr;   // set per launch before conv3d calls

static void conv3d(hipStream_t st, const _Float16* in, _Float16* out,
                   const float* w, const float* bias, int cin, int cout,
                   int iD, int iH, int iW, int oD, int oH, int oW,
                   int kd, int kh, int kw, int sd, int sh, int sw,
                   int pd, int ph, int pw, int dd, int dh, int dw,
                   int act, int inCtot, int inChOff, int outCtot, int outChOff,
                   const _Float16* res = nullptr, float resScale = 1.0f,
                   float outScale = 1.0f, float* outf = nullptr,
                   const float* resf = nullptr) {
  ConvP p;
  p.in = in; p.w = w; p.bias = bias; p.res = res; p.resf = resf;
  p.zp = g_zp; p.out = out; p.outf = outf;
  p.cin = cin; p.cout = cout;
  p.inD = iD; p.inH = iH; p.inW = iW;
  p.outD = oD; p.outH = oH; p.outW = oW;
  p.kd = kd; p.kh = kh; p.kw = kw;
  p.sd = sd; p.sh = sh; p.sw = sw;
  p.pd = pd; p.ph = ph; p.pw = pw;
  p.dd = dd; p.dh = dh; p.dw = dw;
  p.act = act;
  p.inCtot = inCtot; p.inChOff = inChOff;
  p.outCtot = outCtot; p.outChOff = outChOff;
  p.resScale = resScale; p.outScale = outScale;
  p.ntaps = kd * kh * kw; p.khw = kh * kw;
  p.npairs = cdiv(p.ntaps, 2);
  p.ntilesW = cdiv(oW, 16);
  p.mbBlocks = cdiv(cout, 16);
  p.cbBlocks = cdiv(cin, 16);
  const int upr = p.ntilesW * p.mbBlocks;
  int rows = 1;
  while (rows < oH && rows * upr < 16) rows++;
  p.rows = rows;
  p.rowBlocks = cdiv(oH, rows);
  p.units = rows * upr;
  const size_t shmem = LDS_A_OFF +
      (size_t)p.mbBlocks * p.cbBlocks * p.npairs * 512 * sizeof(_Float16);
  conv3d_wmma_kernel<<<oD * p.rowBlocks, 128, shmem, st>>>(p);
}

extern "C" void kernel_launch(void* const* d_in, const int* in_sizes, int n_in,
                              void* d_out, int out_size, void* d_ws, size_t ws_size,
                              hipStream_t stream) {
  (void)in_sizes; (void)n_in; (void)out_size; (void)ws_size;
  const int D = 31, H = 96, W = 96;
  const int HW = H * W;            // 9216
  const int V = D * HW;            // 285696
  auto IN = [&](int i) { return (const float*)d_in[i]; };

  const float *dec_b[3] = { IN(0), IN(2), IN(4) };
  const float *dec_w[3] = { IN(1), IN(3), IN(5) };
  auto ENC = [&](int blk, int k) { return IN(6 + 16 * blk + k); };
  const float *feat_b = IN(54),  *feat_w = IN(55);
  const float *nconv_b = IN(56), *nconv_w = IN(57);
  const float *recon_b = IN(58), *recon_w = IN(59);
  const float *s1b[4] = { IN(60), IN(62), IN(64), IN(66) };
  const float *s1w[4] = { IN(61), IN(63), IN(65), IN(67) };
  const float *s2b[4] = { IN(68), IN(70), IN(72), IN(74) };
  const float *s2w[4] = { IN(69), IN(71), IN(73), IN(75) };
  const float *s3b[3] = { IN(76), IN(78), IN(80) };
  const float *s3w[3] = { IN(77), IN(79), IN(81) };
  const float *x = IN(82);
  float* dout = (float*)d_out;

  // f16 workspace, units of V halfs (total ~194V halfs ~= 111 MB)
  _Float16* B0 = (_Float16*)d_ws;
  auto R = [&](int off) { return B0 + (size_t)off * V; };
  _Float16* XH  = R(0);     // 16V : x cast to f16, padded 16ch
  _Float16* G   = R(16);    // 32V : spec1 gates | B: HD2,HS2 | D: XS0,XS1
  _Float16* T1  = R(48);    // 16V : tmp | B: G2 | D: XS2
  _Float16* T2  = R(64);    // 16V : tmp | B: SMa | D: NX
  _Float16* HD1 = R(80);    // 16V : hd1/q2in | decoder PING
  _Float16* HS1 = R(96);    // 16V : hs1 | B: SMb
  _Float16* O   = R(112);   // 16V : main feature buffer
  _Float16* CAT = R(128);   // 32V : bootstrap concat | C: TCA
  _Float16* CB  = R(160);   // 16V : convb inner | C: TCI
  _Float16* X1O = R(176);   // 16V : x1 out | C: TCB/TCC/TCD/NE
  _Float16* NF  = R(192);   // noise_f [31][16]
  _Float16* ZP  = R(193);   // 32B zero page
  _Float16* XS0 = G, *XS1 = R(32), *XS2 = T1, *NX = T2, *PING = HD1;
  _Float16* SMa = T2, *SMb = HS1, *G2 = T1, *HD2 = G, *HS2 = R(32);
  _Float16* TCI = CB;
  _Float16* TCA = CAT;                          // [31,48,48,16]
  _Float16* TCB = X1O;                          // [31,16,16,16]
  _Float16* TCC = TCB + (size_t)31 * 256 * 16;  // [31,8,8,16]
  _Float16* TCD = TCC + (size_t)31 * 64 * 16;   // [31,16]
  _Float16* NE  = TCD + 31 * 16;                // [31,16]
  g_zp = ZP;

  const int EB = 256;
  auto ew = [&](int n) { return dim3(cdiv(n, EB)); };

  cast_x_kernel<<<ew(V), EB, 0, stream>>>(x, XH, ZP, V);

  // ---- Stage A: qrnn(x, spec1, hidden=16) ----
  conv3d(stream, XH, T1, s1w[0], s1b[0], 1,16, D,H,W, D,H,W, 1,3,3, 1,1,1, 0,1,1, 1,1,1, ACT_NONE, 16,0, 16,0);
  conv3d(stream, T1, T2, s1w[1], s1b[1], 16,16, D,H,W, D,H,W, 3,1,1, 1,1,1, 1,0,0, 1,1,1, ACT_NONE, 16,0, 16,0);
  conv3d(stream, T2, T1, s1w[2], s1b[2], 16,16, D,H,W, D,H,W, 1,3,3, 1,1,1, 0,2,2, 1,2,2, ACT_NONE, 16,0, 16,0);
  conv3d(stream, T1, G,  s1w[3], s1b[3], 16,32, D,H,W, D,H,W, 3,1,1, 1,1,1, 2,0,0, 2,1,1, ACT_LRELU, 16,0, 32,0);
  zf_kernel<<<ew(V * 32), EB, 0, stream>>>(G, 16, 32, V);
  qrnn_scan_kernel<<<ew(16 * HW), EB, 0, stream>>>(G, HD1, HS1, 16, D, HW, 32, 16);
  axpy_kernel<<<ew(16 * V), EB, 0, stream>>>(HD1, HD1, HS1, 2.0f, 16 * V);

  // ---- Stage B: qrnn(q2in, spec2, hidden=1) ----
  conv3d(stream, HD1, SMa, s2w[0], s2b[0], 16,1, D,H,W, D,H,W, 1,3,3, 1,1,1, 0,1,1, 1,1,1, ACT_NONE, 16,0, 16,0);
  conv3d(stream, SMa, SMb, s2w[1], s2b[1], 1,1,  D,H,W, D,H,W, 3,1,1, 1,1,1, 1,0,0, 1,1,1, ACT_NONE, 16,0, 16,0);
  conv3d(stream, SMb, SMa, s2w[2], s2b[2], 1,1,  D,H,W, D,H,W, 1,3,3, 1,1,1, 0,2,2, 1,2,2, ACT_NONE, 16,0, 16,0);
  conv3d(stream, SMa, G2,  s2w[3], s2b[3], 1,2,  D,H,W, D,H,W, 3,1,1, 1,1,1, 2,0,0, 2,1,1, ACT_LRELU, 16,0, 16,0);
  zf_kernel<<<ew(V * 2), EB, 0, stream>>>(G2, 1, 16, V);
  qrnn_scan_kernel<<<ew(16 * HW), EB, 0, stream>>>(G2, HD2, HS2, 1, D, HW, 16, 16);
  axpy_kernel<<<ew(16 * V), EB, 0, stream>>>(TCI, HD2, HS2, 2.0f, 16 * V);

  // ---- Stage C: transconvb -> noise_f ----
  conv3d(stream, TCI, TCA, s3w[0], s3b[0], 1,8, D,96,96, D,48,48, 1,5,5, 1,2,2, 0,2,2, 1,1,1, ACT_RELU, 16,0, 16,0);
  blockmean_kernel<<<ew(31*16*16*16), EB, 0, stream>>>(TCA, TCB, 48,48, 16,16, 3,3, 31*16*16*16);
  conv3d(stream, TCB, TCC, s3w[1], s3b[1], 8,4, D,16,16, D,8,8, 1,3,3, 1,2,2, 0,1,1, 1,1,1, ACT_RELU, 16,0, 16,0);
  meanhw_kernel<<<ew(31 * 16), EB, 0, stream>>>(TCC, TCD, 31, 64);
  conv3d(stream, TCD, NE, s3w[2], s3b[2], 4,1, D,1,1, D,1,1, 1,1,1, 1,1,1, 0,0,0, 1,1,1, ACT_SIGMOID, 16,0, 16,0);
  conv3d(stream, NE,  NF, nconv_w, nconv_b, 1,16, D,1,1, D,1,1, 1,1,1, 1,1,1, 0,0,0, 1,1,1, ACT_NONE, 16,0, 16,0);

  // ---- Stage D: feature path ----
  conv3d(stream, XH, O, feat_w, feat_b, 1,16, D,H,W, D,H,W, 3,3,3, 1,1,1, 1,1,1, 1,1,1, ACT_NONE, 16,0, 16,0);

  _Float16* XS[3] = { XS0, XS1, XS2 };
  for (int e = 0; e < 3; ++e) {
    hipMemcpyAsync(XS[e], O, (size_t)16 * V * sizeof(_Float16),
                   hipMemcpyDeviceToDevice, stream);
    mulnoise_kernel<<<ew(16 * V), EB, 0, stream>>>(NX, O, NF, HW, 16 * V);
    // p1: convb(nx, k=5, pad 2) -> xa = CAT ch0..7
    conv3d(stream, NX,  CB,  ENC(e,1), ENC(e,0), 16,4, D,H,W, D,H,W, 1,5,5, 1,1,1, 0,2,2, 1,1,1, ACT_NONE, 16,0, 16,0);
    conv3d(stream, CB,  CAT, ENC(e,3), ENC(e,2), 4,8,  D,H,W, D,H,W, 5,1,1, 1,1,1, 2,0,0, 1,1,1, ACT_LRELU, 16,0, 32,0);
    // p2: convb(xa, k=3, pad 1) -> xb = CAT ch8..15
    conv3d(stream, CAT, CB,  ENC(e,5), ENC(e,4), 8,4,  D,H,W, D,H,W, 1,3,3, 1,1,1, 0,1,1, 1,1,1, ACT_NONE, 32,0, 16,0);
    conv3d(stream, CB,  CAT, ENC(e,7), ENC(e,6), 4,8,  D,H,W, D,H,W, 3,1,1, 1,1,1, 1,0,0, 1,1,1, ACT_LRELU, 16,0, 32,8);
    // p3: convb(xb, k=1) -> xc = CAT ch16..23
    conv3d(stream, CAT, CB,  ENC(e,9), ENC(e,8), 8,4,  D,H,W, D,H,W, 1,1,1, 1,1,1, 0,0,0, 1,1,1, ACT_NONE, 32,8, 16,0);
    conv3d(stream, CB,  CAT, ENC(e,11), ENC(e,10), 4,8, D,H,W, D,H,W, 1,1,1, 1,1,1, 0,0,0, 1,1,1, ACT_LRELU, 16,0, 32,16);
    // x1 on concat(24ch) then x2 1x1x1 + residual(O), in place
    conv3d(stream, CAT, X1O, ENC(e,13), ENC(e,12), 24,8, D,H,W, D,H,W, 3,3,3, 1,1,1, 1,1,1, 1,1,1, ACT_LRELU, 32,0, 16,0);
    conv3d(stream, X1O, O,   ENC(e,15), ENC(e,14), 8,16, D,H,W, D,H,W, 1,1,1, 1,1,1, 0,0,0, 1,1,1, ACT_NONE, 16,0, 16,0,
           O, 1.0f, 1.0f);
  }

  // ---- decoder: out = lrelu(conv(out)) + skip ----
  conv3d(stream, O,    PING, dec_w[0], dec_b[0], 16,16, D,H,W, D,H,W, 3,3,3, 1,1,1, 1,1,1, 1,1,1, ACT_LRELU, 16,0, 16,0, XS2, 1.0f, 1.0f);
  conv3d(stream, PING, O,    dec_w[1], dec_b[1], 16,16, D,H,W, D,H,W, 3,3,3, 1,1,1, 1,1,1, 1,1,1, ACT_LRELU, 16,0, 16,0, XS1, 1.0f, 1.0f);
  conv3d(stream, O,    PING, dec_w[2], dec_b[2], 16,16, D,H,W, D,H,W, 3,3,3, 1,1,1, 1,1,1, 1,1,1, ACT_LRELU, 16,0, 16,0, XS0, 1.0f, 1.0f);

  // ---- recon + final subtraction: d_out = x - conv(PING, recon) ----
  conv3d(stream, PING, nullptr, recon_w, recon_b, 16,1, D,H,W, D,H,W, 3,3,3, 1,1,1, 1,1,1, 1,1,1, ACT_NONE, 16,0, 16,0,
         nullptr, 1.0f, -1.0f, dout, x);
}
